// GraphEmbt_7791070675733
// MI455X (gfx1250) — matrix operand order repaired and checked by reading, not compiled, not run on hardware
//
#include <hip/hip_runtime.h>

typedef _Float16 v16h __attribute__((ext_vector_type(16)));
typedef _Float16 v8h  __attribute__((ext_vector_type(8)));
typedef float    v8f  __attribute__((ext_vector_type(8)));

#define N_IMAGE 64
#define N_CAP   64
#define N_REG   36
#define RP      48      /* regions padded to 3x16 */
#define RP2     64      /* regions padded to 2x32 (K for stage C) */
#define MAXW    40
#define LP      48      /* words padded to 3x16 */
#define EMBED   1024
#define SIMDIM  256
#define F_EPS   1e-8f
#define SMOOTH  9.0f

// workspace layout (units of _Float16)
#define WT_OFF   ((size_t)0)                                  // W^T : [256][1024]
#define IMGA_OFF ((size_t)(256*1024))                         // img : [64][48][1024]
#define IMGT_OFF (IMGA_OFF + (size_t)64*RP*EMBED)             // img^T: [64][1024][64]
#define CAPB_OFF (IMGT_OFF + (size_t)64*EMBED*RP2)            // cap : [64][48][1024]

#define SHUF16(lo, hi) __builtin_shufflevector(lo, hi, 0,1,2,3,4,5,6,7,8,9,10,11,12,13,14,15)

// CDNA5 async DMA: global -> LDS, 16B per lane, tracked by ASYNCcnt
__device__ __forceinline__ void async_copy16(unsigned lds_off, const void* gptr) {
  asm volatile("global_load_async_to_lds_b128 %0, %1, off"
               :: "v"(lds_off), "v"(gptr) : "memory");
}
__device__ __forceinline__ void wait_async0() {
  asm volatile("s_wait_asynccnt 0x0" ::: "memory");
}

// ---------------- prep kernels: f32 -> f16, padded & transposed ----------------
__global__ void prep_w(const float* __restrict__ W, _Float16* __restrict__ wt) {
  int idx = blockIdx.x * 256 + threadIdx.x;          // s*1024 + d
  int s = idx >> 10, d = idx & 1023;
  wt[idx] = (_Float16)W[d * SIMDIM + s];
}
__global__ void prep_imgA(const float* __restrict__ img, _Float16* __restrict__ dst) {
  int idx = blockIdx.x * 256 + threadIdx.x;          // (i*48 + r)*1024 + d
  int d = idx & 1023; int r = (idx >> 10) % RP; int i = idx / (RP * 1024);
  float v = (r < N_REG) ? img[((size_t)i * N_REG + r) * EMBED + d] : 0.f;
  dst[idx] = (_Float16)v;
}
__global__ void prep_imgT(const float* __restrict__ img, _Float16* __restrict__ dst) {
  int idx = blockIdx.x * 256 + threadIdx.x;          // (i*1024 + d)*64 + r
  int r = idx & 63; int d = (idx >> 6) & 1023; int i = idx >> 16;
  float v = (r < N_REG) ? img[((size_t)i * N_REG + r) * EMBED + d] : 0.f;
  dst[idx] = (_Float16)v;
}
__global__ void prep_capB(const float* __restrict__ cap, _Float16* __restrict__ dst) {
  int idx = blockIdx.x * 256 + threadIdx.x;          // (c*48 + l)*1024 + d
  int d = idx & 1023; int l = (idx >> 10) % LP; int i = idx / (LP * 1024);
  float v = (l < MAXW) ? cap[((size_t)i * MAXW + l) * EMBED + d] : 0.f;
  dst[idx] = (_Float16)v;
}

// ---------------- fused per-(caption,image) kernel ----------------
__launch_bounds__(256)
__global__ void scan_sim_kernel(const _Float16* __restrict__ wt,
                                const _Float16* __restrict__ imgA,
                                const _Float16* __restrict__ imgT,
                                const _Float16* __restrict__ capB,
                                const float*    __restrict__ capF,
                                const float*    __restrict__ bias,
                                const int*      __restrict__ cap_lens,
                                float*          __restrict__ out) {
  // 48KB region: stage A ping-pong staging (2 x 24KB) aliased with stage C/D slabs
  __shared__ __align__(32) unsigned char s_union[49152];
  __shared__ float s_attn[RP * LP];                       // [r][l] f32
  __shared__ __align__(32) _Float16 s_attn_h[LP * RP2];   // softmaxed [l][64] f16
  __shared__ float s_invn[16];
  __shared__ float s_onorm[16];

  _Float16* s_wc  = (_Float16*)s_union;                   // [16][1024] wc / sim slab
  float*    s_out = (float*)(s_union + 32768);            // [16][256]

  const int tid  = threadIdx.x;
  const int wave = tid >> 5;
  const int lane = tid & 31;
  const int mlan = lane & 15;
  const int hh   = lane >> 4;            // lane half selects K sub-block
  const int c = blockIdx.x >> 6;
  const int i = blockIdx.x & 63;
  const unsigned lds_u = (unsigned)(uintptr_t)(void*)s_union;

  // warm L2 with W^T (resident across all 4096 blocks)
  __builtin_prefetch(wt + (size_t)wave * 32768, 0, 1);

  // ===== Stage A: attn[r][l] = sum_d img[i][r][d] * cap[c][l][d]
  // K staged through LDS in 128-half chunks, ping-pong async DMA overlapped with WMMA.
  const int mt0 = wave / 3, nt0 = wave % 3;               // tile owned by this wave
  v8f accT0 = {}, accT1 = {};

  auto issue_chunk = [&](int ch) {
    const int kb0 = ch * 128;
    const unsigned base = lds_u + (unsigned)(ch & 1) * 24576u;
#pragma unroll
    for (int q = 0; q < 3; ++q) {                         // 2 x 12288B slabs, 16B units
      int u = tid + q * 256;                              // 0..767
      int row = u >> 4;                                   // 256B (16 units) per row
      int col = (u & 15) * 8;                             // halves within row
      async_copy16(base + u * 16,
                   imgA + ((size_t)i * RP + row) * EMBED + kb0 + col);
      async_copy16(base + 12288u + u * 16,
                   capB + ((size_t)c * LP + row) * EMBED + kb0 + col);
    }
  };

  issue_chunk(0);
  for (int ch = 0; ch < 8; ++ch) {
    wait_async0();                                        // own chunk-ch writes landed
    __syncthreads();                                      // all waves' writes visible;
                                                          // other buffer fully consumed
    if (ch + 1 < 8) issue_chunk(ch + 1);                  // DMA overlaps compute below
    const _Float16* buf = (const _Float16*)(s_union + (ch & 1) * 24576);
    {
      const _Float16* aB = buf + (mt0 * 16 + mlan) * 128;
      const _Float16* bB = buf + 6144 + (nt0 * 16 + mlan) * 128;
#pragma unroll
      for (int k8 = 0; k8 < 128; k8 += 32) {
        v8h alo = *(const v8h*)(aB + k8 + hh * 16);
        v8h ahi = *(const v8h*)(aB + k8 + hh * 16 + 8);
        v16h a = SHUF16(alo, ahi);
        v16h b = *(const v16h*)(bB + k8 + hh * 16);
        accT0 = __builtin_amdgcn_wmma_f32_16x16x32_f16(false, a, false, b, (short)0, accT0, false, false);
      }
      if (wave == 0) {                                    // 9th tile (mt=2, nt=2)
        const _Float16* aB2 = buf + (32 + mlan) * 128;
        const _Float16* bB2 = buf + 6144 + (32 + mlan) * 128;
#pragma unroll
        for (int k8 = 0; k8 < 128; k8 += 32) {
          v8h alo = *(const v8h*)(aB2 + k8 + hh * 16);
          v8h ahi = *(const v8h*)(aB2 + k8 + hh * 16 + 8);
          v16h a = SHUF16(alo, ahi);
          v16h b = *(const v16h*)(bB2 + k8 + hh * 16);
          accT1 = __builtin_amdgcn_wmma_f32_16x16x32_f16(false, a, false, b, (short)0, accT1, false, false);
        }
      }
    }
  }
  __syncthreads();                                        // staging reads done
#pragma unroll
  for (int j = 0; j < 8; ++j)
    s_attn[(mt0 * 16 + hh * 8 + j) * LP + nt0 * 16 + mlan] = accT0[j];
  if (wave == 0) {
#pragma unroll
    for (int j = 0; j < 8; ++j)
      s_attn[(32 + hh * 8 + j) * LP + 32 + mlan] = accT1[j];
  }
  __syncthreads();

  // ===== Stage B1: leaky-relu + l2norm over words (per region row) =====
  if (tid < RP) {
    int r = tid;
    float nrm = 0.f;
    for (int l = 0; l < MAXW; ++l) {
      float v = s_attn[r * LP + l];
      v = (v > 0.f) ? v : 0.1f * v;
      s_attn[r * LP + l] = v;
      nrm += v * v;
    }
    float inv = 1.f / (sqrtf(nrm) + F_EPS);
    for (int l = 0; l < MAXW; ++l) s_attn[r * LP + l] *= inv;
  }
  __syncthreads();

  // ===== Stage B2: softmax over regions (x9), emit f16 A-matrix [l][64] =====
  if (tid < LP) {
    int l = tid;
    if (l < MAXW) {
      float mx = -3.4e38f;
      for (int r = 0; r < N_REG; ++r) mx = fmaxf(mx, SMOOTH * s_attn[r * LP + l]);
      float sum = 0.f;
      for (int r = 0; r < N_REG; ++r) sum += __expf(SMOOTH * s_attn[r * LP + l] - mx);
      float rinv = 1.f / sum;
      for (int r = 0; r < N_REG; ++r)
        s_attn_h[l * RP2 + r] = (_Float16)(__expf(SMOOTH * s_attn[r * LP + l] - mx) * rinv);
      for (int r = N_REG; r < RP2; ++r) s_attn_h[l * RP2 + r] = (_Float16)0.f;
    } else {
      for (int r = 0; r < RP2; ++r) s_attn_h[l * RP2 + r] = (_Float16)0.f;
    }
  }

  // ===== Stages C+D tiled over three 16-word slabs =====
  for (int mt = 0; mt < 3; ++mt) {
    __syncthreads();   // s_attn_h ready; previous slab fully consumed

    // ---- Stage C: wc[l][d] = sum_r attn_sm[l][r] * img[i][r][d]  (K=64 padded)
    for (int nt = wave * 8; nt < wave * 8 + 8; ++nt) {
      const _Float16* bB = imgT + ((size_t)i * EMBED + nt * 16 + mlan) * RP2 + hh * 16;
      v16h b0 = *(const v16h*)(bB);
      v16h b1 = *(const v16h*)(bB + 32);
      const _Float16* aB = &s_attn_h[(mt * 16 + mlan) * RP2];
      v8h alo0 = *(const v8h*)(aB + hh * 16);
      v8h ahi0 = *(const v8h*)(aB + hh * 16 + 8);
      v8h alo1 = *(const v8h*)(aB + 32 + hh * 16);
      v8h ahi1 = *(const v8h*)(aB + 32 + hh * 16 + 8);
      v16h a0 = SHUF16(alo0, ahi0);
      v16h a1 = SHUF16(alo1, ahi1);
      v8f acc = {};
      acc = __builtin_amdgcn_wmma_f32_16x16x32_f16(false, a0, false, b0, (short)0, acc, false, false);
      acc = __builtin_amdgcn_wmma_f32_16x16x32_f16(false, a1, false, b1, (short)0, acc, false, false);
#pragma unroll
      for (int j = 0; j < 8; ++j)
        s_wc[(hh * 8 + j) * EMBED + nt * 16 + mlan] = (_Float16)acc[j];
    }
    __syncthreads();

    // ---- wc row norms (wave32 shuffle reduction)
    for (int lr = wave; lr < 16; lr += 8) {
      float sum = 0.f;
      for (int d = lane; d < EMBED; d += 32) {
        float v = (float)s_wc[lr * EMBED + d];
        sum += v * v;
      }
#pragma unroll
      for (int off = 16; off > 0; off >>= 1) sum += __shfl_xor(sum, off, 32);
      if (lane == 0) {
        int l = mt * 16 + lr;
        s_invn[lr] = (l < MAXW) ? 1.f / (sqrtf(sum) + F_EPS) : 0.f;
      }
    }
    if (tid < 16) s_onorm[tid] = 0.f;
    __syncthreads();

    // ---- in-place transform: s_wc <- sim = (wc*invn - cap)^2, vectorized
    {
      const int m = tid >> 4;
      const int col0 = (tid & 15) * 64;
      const int l = mt * 16 + m;
      const bool valid = (l < MAXW);
      const float invn = s_invn[m];
      const float* capRow = capF + ((size_t)c * MAXW + (valid ? l : 0)) * EMBED + col0;
#pragma unroll
      for (int u = 0; u < 64; u += 8) {
        v8h wv = *(v8h*)(s_wc + m * EMBED + col0 + u);
#pragma unroll
        for (int e = 0; e < 8; ++e) {
          float w = (float)wv[e] * invn;
          float cv = valid ? capRow[u + e] : 0.f;
          float dd = w - cv;
          wv[e] = (_Float16)(dd * dd);
        }
        *(v8h*)(s_wc + m * EMBED + col0 + u) = wv;
      }
    }
    __syncthreads();

    // ---- Stage D: out = relu(sim @ W + b); W-stream pipelined 2 K-steps deep
    {
      const int st0 = wave * 2;
      const _Float16* w0 = wt + ((size_t)(st0 * 16 + mlan)) * EMBED + hh * 16;
      const _Float16* w1 = wt + ((size_t)(st0 * 16 + 16 + mlan)) * EMBED + hh * 16;
      v8f acc0 = {}, acc1 = {};
      v16h b0a = *(const v16h*)(w0);
      v16h b1a = *(const v16h*)(w1);
      v16h b0b = *(const v16h*)(w0 + 32);
      v16h b1b = *(const v16h*)(w1 + 32);
      for (int kb = 0; kb < EMBED; kb += 32) {
        const int kn = (kb + 64 < EMBED) ? (kb + 64) : kb;     // 2-step lookahead
        v16h b0n = *(const v16h*)(w0 + kn);
        v16h b1n = *(const v16h*)(w1 + kn);
        v8h alo = *(const v8h*)(s_wc + mlan * EMBED + kb + hh * 16);
        v8h ahi = *(const v8h*)(s_wc + mlan * EMBED + kb + hh * 16 + 8);
        v16h a = SHUF16(alo, ahi);
        acc0 = __builtin_amdgcn_wmma_f32_16x16x32_f16(false, a, false, b0a, (short)0, acc0, false, false);
        acc1 = __builtin_amdgcn_wmma_f32_16x16x32_f16(false, a, false, b1a, (short)0, acc1, false, false);
        b0a = b0b; b1a = b1b;
        b0b = b0n; b1b = b1n;
      }
#pragma unroll
      for (int tsel = 0; tsel < 2; ++tsel) {
        v8f acc = tsel ? acc1 : acc0;
        int s = (st0 + tsel) * 16 + mlan;
        float bs = bias[s];
#pragma unroll
        for (int j = 0; j < 8; ++j) {
          int m = hh * 8 + j;
          float v = fmaxf(acc[j] + bs, 0.f);
          s_out[m * SIMDIM + s] = v;
          atomicAdd(&s_onorm[m], v * v);
        }
      }
    }
    __syncthreads();

    // ---- final l2norm + word-mask + store
    {
      const int cl = cap_lens[c];
      for (int e = tid; e < 16 * SIMDIM; e += 256) {
        int m = e >> 8, s = e & 255;
        int l = mt * 16 + m;
        if (l < MAXW) {
          float sc = (l < cl) ? 1.f / (sqrtf(s_onorm[m]) + F_EPS) : 0.f;
          out[(((size_t)(c * N_IMAGE + i)) * MAXW + l) * SIMDIM + s] = s_out[m * SIMDIM + s] * sc;
        }
      }
    }
  }
}

extern "C" void kernel_launch(void* const* d_in, const int* in_sizes, int n_in,
                              void* d_out, int out_size, void* d_ws, size_t ws_size,
                              hipStream_t stream) {
  const float* img  = (const float*)d_in[0];
  const float* cap  = (const float*)d_in[1];
  const int*   lens = (const int*)d_in[2];
  const float* W    = (const float*)d_in[5];
  const float* b    = (const float*)d_in[6];

  _Float16* ws   = (_Float16*)d_ws;
  _Float16* wt   = ws + WT_OFF;
  _Float16* imgA = ws + IMGA_OFF;
  _Float16* imgT = ws + IMGT_OFF;
  _Float16* capB = ws + CAPB_OFF;

  prep_w   <<< (256 * 1024) / 256,       256, 0, stream >>>(W,   wt);
  prep_imgA<<< (64 * RP * EMBED) / 256,  256, 0, stream >>>(img, imgA);
  prep_imgT<<< (64 * EMBED * RP2) / 256, 256, 0, stream >>>(img, imgT);
  prep_capB<<< (64 * LP * EMBED) / 256,  256, 0, stream >>>(cap, capB);

  scan_sim_kernel<<< N_CAP * N_IMAGE, 256, 0, stream >>>(
      wt, imgA, imgT, capB, cap, b, lens, (float*)d_out);
}
